// TEPGNN_36541581754570
// MI455X (gfx1250) — matrix operand order repaired and checked by reading, not compiled, not run on hardware
//
#include <hip/hip_runtime.h>
#include <cstdint>

#define N_NODES  100000
#define N_EDGES  3200000
#define N_GRAPHS 1024
#define HID      64
#define NCLS     21

typedef __attribute__((ext_vector_type(16))) _Float16 v16h;
typedef __attribute__((ext_vector_type(8)))  float    v8f;

static __device__ __forceinline__ void atomAddF(float* p, float v) {
    __hip_atomic_fetch_add(p, v, __ATOMIC_RELAXED, __HIP_MEMORY_SCOPE_AGENT);
}

// ---------------- Layer 1: per-edge scalar aggregation ----------------
__global__ void k_edge1(const long long* __restrict__ ei,
                        const float* __restrict__ x,
                        float* __restrict__ sum1, float* __restrict__ cnt) {
    int e = blockIdx.x * blockDim.x + threadIdx.x;
    if (e >= N_EDGES) return;
    int s = (int)ei[e];
    int d = (int)ei[N_EDGES + e];
    atomAddF(&sum1[d], x[s]);
    atomAddF(&cnt[d], 1.0f);
}

// h1[n,f] = relu(mean_agg(n)*W1l[f] + b1[f] + x[n]*W1r[f])   (fan_in == 1)
__global__ void k_h1(const float* __restrict__ x, const float* __restrict__ sum1,
                     const float* __restrict__ cnt,
                     const float* __restrict__ W1l, const float* __restrict__ b1,
                     const float* __restrict__ W1r, float* __restrict__ h1) {
    int t = blockIdx.x * blockDim.x + threadIdx.x;
    if (t >= N_NODES * HID) return;
    int n = t >> 6, f = t & 63;
    float c = cnt[n]; c = c > 1.0f ? c : 1.0f;
    float agg = sum1[n] / c;
    float v = agg * W1l[f] + b1[f] + x[n] * W1r[f];
    h1[t] = v > 0.0f ? v : 0.0f;
}

// ------------- Pack [W2l ; W2r] (128 x 64) into f16 once per call -------------
__global__ void k_wcat(const float* __restrict__ W2l, const float* __restrict__ W2r,
                       _Float16* __restrict__ Wcat) {
    int t = blockIdx.x * blockDim.x + threadIdx.x;
    if (t >= 128 * HID) return;
    int r = t >> 6, c = t & 63;
    float v = (r < 64) ? W2l[r * HID + c] : W2r[(r - 64) * HID + c];
    Wcat[t] = (_Float16)v;
}

// ---------------- Layer 2: per-edge 64-wide scatter (1 wave / edge) ----------------
__global__ void k_edge2(const long long* __restrict__ ei,
                        const float* __restrict__ h1, float* __restrict__ agg2) {
    long long t = (long long)blockIdx.x * blockDim.x + threadIdx.x;
    long long e = t >> 5;
    if (e >= N_EDGES) return;
    int lane = threadIdx.x & 31;
    __builtin_prefetch(ei + e + 2048, 0, 0);       // index-stream prefetch
    int s = (int)ei[e];
    int d = (int)ei[N_EDGES + e];
    float2 v = ((const float2*)(h1 + (long long)s * HID))[lane];
    float* o = agg2 + (long long)d * HID + lane * 2;
    atomAddF(o,     v.x);
    atomAddF(o + 1, v.y);
}

// ----- Fused SAGE-2 GEMM via WMMA: h2 = relu([agg2/cnt | h1] @ [W2l;W2r] + b2) -----
// One wave computes a 16x64 output tile: K=128 as 4 chunks of 32, N=64 as 4 tiles of 16.
// Weights staged once per workgroup into LDS; B fragments are gathered from LDS one f16 at a time.
__global__ void __launch_bounds__(256)
k_sage2(const float* __restrict__ agg2, const float* __restrict__ cnt,
        const float* __restrict__ h1, const _Float16* __restrict__ Wcat,
        const float* __restrict__ b2, float* __restrict__ h2) {
    __shared__ __align__(32) _Float16 sW[128 * HID];   // 16 KB of 320 KB/WGP

    // cooperative stage: 1024 x uint4 (16 KB), 256 threads x 4 each
    {
        const uint4* gsrc = (const uint4*)Wcat;
        uint4* ldst = (uint4*)sW;
        int t = threadIdx.x;
#pragma unroll
        for (int i = 0; i < 4; ++i)
            ldst[t + i * 256] = gsrc[t + i * 256];
    }
    __syncthreads();   // before the wave guard: every wave reaches the barrier

    int wave = (blockIdx.x * blockDim.x + threadIdx.x) >> 5;   // tile id (uniform/wave)
    if (wave >= N_NODES / 16) return;                          // EXEC stays all-ones
    int lane = threadIdx.x & 31;
    int g    = lane >> 4;          // half-group: selects K+8 per ISA A-layout
    int mrow = lane & 15;          // M row this lane feeds for the A fragment
    long long row = (long long)wave * 16 + mrow;

    float c = cnt[row]; c = c > 1.0f ? c : 1.0f;
    float invc = 1.0f / c;

    v8f acc[4] = {};               // N-tiles 0..3 (cols 0-15,16-31,32-47,48-63)

#pragma unroll
    for (int kc = 0; kc < 4; ++kc) {
        const float* src = (kc < 2) ? agg2 : h1;       // K 0-63: agg, K 64-127: h1
        float scale      = (kc < 2) ? invc : 1.0f;     // fold the mean-divide into A
        int kbase        = (kc & 1) * 32;

        // A fragment (16x32 f16): halves 0-7 -> K = kbase+8g+h, halves 8-15 -> K+16
        const float4* p0 = (const float4*)(src + row * HID + kbase + 8 * g);
        const float4* p1 = (const float4*)(src + row * HID + kbase + 16 + 8 * g);
        float4 A0 = p0[0], A1 = p0[1], A2 = p1[0], A3 = p1[1];
        float av[16] = {A0.x, A0.y, A0.z, A0.w, A1.x, A1.y, A1.z, A1.w,
                        A2.x, A2.y, A2.z, A2.w, A3.x, A3.y, A3.z, A3.w};
        v16h a;
#pragma unroll
        for (int i = 0; i < 16; ++i) a[i] = (_Float16)(av[i] * scale);

        // B fragments (32x16 f16) from LDS: N = lane%16, halves 0-7 -> K = 8g+h, halves 8-15 -> K+16
        v16h b0, b1, b2v, b3;
#pragma unroll
        for (int i = 0; i < 16; ++i) {
            int kk = kc * 32 + ((i < 8) ? (8 * g + i) : (16 + 8 * g + (i - 8)));
            const _Float16* wrow = sW + kk * HID + mrow;
            b0[i]  = wrow[0];
            b1[i]  = wrow[16];
            b2v[i] = wrow[32];
            b3[i]  = wrow[48];
        }

        acc[0] = __builtin_amdgcn_wmma_f32_16x16x32_f16(false, a, false, b0, (short)0, acc[0], false, false);
        acc[1] = __builtin_amdgcn_wmma_f32_16x16x32_f16(false, a, false, b1, (short)0, acc[1], false, false);
        acc[2] = __builtin_amdgcn_wmma_f32_16x16x32_f16(false, a, false, b2v,(short)0, acc[2], false, false);
        acc[3] = __builtin_amdgcn_wmma_f32_16x16x32_f16(false, a, false, b3, (short)0, acc[3], false, false);
    }

    // C/D layout: lane -> N = lane%16, VGPR v -> M = v + 8*(lane/16)
    int coln = lane & 15;
#pragma unroll
    for (int nt = 0; nt < 4; ++nt) {
        float bias = b2[nt * 16 + coln];
#pragma unroll
        for (int v = 0; v < 8; ++v) {
            long long r = (long long)wave * 16 + v + 8 * g;
            float val = acc[nt][v] + bias;
            h2[r * HID + nt * 16 + coln] = val > 0.0f ? val : 0.0f;
        }
    }
}

// ---------------- Global mean pool ----------------
__global__ void k_pool(const long long* __restrict__ batch, const float* __restrict__ h2,
                       float* __restrict__ pooled, float* __restrict__ gcnt) {
    int t = blockIdx.x * blockDim.x + threadIdx.x;
    if (t >= N_NODES * HID) return;
    int n = t >> 6, f = t & 63;
    int b = (int)batch[n];
    atomAddF(&pooled[b * HID + f], h2[t]);
    if (f == 0) atomAddF(&gcnt[b], 1.0f);
}

// ---------------- Classifier: 1024 x 64 @ 64 x 21 ----------------
__global__ void k_cls(const float* __restrict__ pooled, const float* __restrict__ gcnt,
                      const float* __restrict__ Wc, const float* __restrict__ bc,
                      float* __restrict__ out) {
    int t = blockIdx.x * blockDim.x + threadIdx.x;
    if (t >= N_GRAPHS * NCLS) return;
    int gph = t / NCLS, c = t % NCLS;
    float cn = gcnt[gph]; cn = cn > 1.0f ? cn : 1.0f;
    float inv = 1.0f / cn;
    float s = 0.0f;
#pragma unroll
    for (int k = 0; k < HID; ++k) s += pooled[gph * HID + k] * Wc[k * NCLS + c];
    out[t] = s * inv + bc[c];
}

extern "C" void kernel_launch(void* const* d_in, const int* in_sizes, int n_in,
                              void* d_out, int out_size, void* d_ws, size_t ws_size,
                              hipStream_t stream) {
    const float*     x   = (const float*)d_in[0];
    const long long* ei  = (const long long*)d_in[1];
    const long long* bat = (const long long*)d_in[2];
    const float* W1l = (const float*)d_in[3];
    const float* b1  = (const float*)d_in[4];
    const float* W1r = (const float*)d_in[5];
    const float* W2l = (const float*)d_in[6];
    const float* b2  = (const float*)d_in[7];
    const float* W2r = (const float*)d_in[8];
    const float* Wc  = (const float*)d_in[9];
    const float* bc  = (const float*)d_in[10];
    float* out = (float*)d_out;

    // bump allocator over d_ws
    char* base = (char*)d_ws;
    size_t off = 0;
    auto alloc = [&](size_t bytes) -> void* {
        off = (off + 255) & ~(size_t)255;
        void* p = base + off;
        off += bytes;
        return p;
    };
    float*    sum1   = (float*)alloc((size_t)N_NODES * 4);
    float*    cnt    = (float*)alloc((size_t)N_NODES * 4);
    float*    h1     = (float*)alloc((size_t)N_NODES * HID * 4);
    float*    agg2   = (float*)alloc((size_t)N_NODES * HID * 4);
    float*    h2     = (float*)alloc((size_t)N_NODES * HID * 4);
    _Float16* Wcat   = (_Float16*)alloc((size_t)128 * HID * 2);
    float*    pooled = (float*)alloc((size_t)N_GRAPHS * HID * 4);
    float*    gcnt   = (float*)alloc((size_t)N_GRAPHS * 4);

    // zero the accumulation buffers (graph-capturable memset nodes)
    hipMemsetAsync(sum1,   0, (size_t)N_NODES * 4,        stream);
    hipMemsetAsync(cnt,    0, (size_t)N_NODES * 4,        stream);
    hipMemsetAsync(agg2,   0, (size_t)N_NODES * HID * 4,  stream);
    hipMemsetAsync(pooled, 0, (size_t)N_GRAPHS * HID * 4, stream);
    hipMemsetAsync(gcnt,   0, (size_t)N_GRAPHS * 4,       stream);

    // layer 1
    k_edge1<<<(N_EDGES + 255) / 256, 256, 0, stream>>>(ei, x, sum1, cnt);
    k_h1<<<(N_NODES * HID + 255) / 256, 256, 0, stream>>>(x, sum1, cnt, W1l, b1, W1r, h1);

    // pack f16 weights for WMMA
    k_wcat<<<(128 * HID + 255) / 256, 256, 0, stream>>>(W2l, W2r, Wcat);

    // layer 2: edge scatter (1 wave per edge) then fused WMMA GEMM
    long long e2threads = (long long)N_EDGES * 32;
    k_edge2<<<(unsigned)((e2threads + 255) / 256), 256, 0, stream>>>(ei, h1, agg2);
    int tiles = N_NODES / 16;                       // 6250, exact
    k_sage2<<<(tiles + 7) / 8, 256, 0, stream>>>(agg2, cnt, h1, Wcat, b2, h2);

    // pool + classify
    k_pool<<<(N_NODES * HID + 255) / 256, 256, 0, stream>>>(bat, h2, pooled, gcnt);
    k_cls<<<(N_GRAPHS * NCLS + 255) / 256, 256, 0, stream>>>(pooled, gcnt, Wc, bc, out);
}
